// PrimitiveEquationsBlockStepper_32152125178574
// MI455X (gfx1250) — matrix operand order repaired and checked by reading, not compiled, not run on hardware
//
// ============================================================================
// PrimitiveEquationsBlockStepper — MI455X (gfx1250, CDNA5) implementation.
//
// The reference collapses (after dropping its all-zero einsum weights) to 80
// applications of a per-latitude 25-tap stencil
//   SG_nc(x)[h,w] = sum_{dl,dw} psiS[nc,h,dl,dw]*quad[li]*x[li,(w+dw-2)%W]
// (psiS = psi summed over kk) + pointwise vertical physics. ~70 MB HBM
// traffic @ 23.3 TB/s ~ 3 us, ~1 GFLOP -> latency/fusion bound.
//
// WMMA mapping (v_wmma_f32_16x16x32_bf16, wave32): one chained WMMA pair
// computes SG_0/SG_1 for EIGHT latitude rows x 16 columns of one field:
//   A rows  m = 2*hr + nc (hr=0..7, nc=0..1)  -> full 16-row M utilization
//   taps    t = (src_row, dw); src rows h-2..h+9 split into two K=30 chunks
//   D       = A2 x B2 + (A1 x B1 + 0)   (accumulator chaining)
// All zero-masking lives in the one-time A build; B-fragment loads are
// unconditional fp32 LDS reads (pad taps hit zero A columns) converted to
// bf16 at pack time. Tile staging rides the CDNA5 async-to-LDS path
// (GLOBAL_LOAD_ASYNC_TO_LDS_B32/B64 + s_wait_asynccnt). Workspace planes are
// padded to 368x768 so stage-1 stores need no guards.
// ============================================================================
#include <hip/hip_runtime.h>

typedef __attribute__((ext_vector_type(16))) __bf16 bf16x16;
typedef __attribute__((ext_vector_type(8)))  float  floatx8;
typedef __attribute__((ext_vector_type(2)))  int    int2v;

#define HGT   361
#define WID   720
#define KLEV  8
#define KKN   5
#define TCOLS 132           // 128-col tile + 2-col halo each side
#define WTILE 128
#define HROWS 8             // latitude rows per block (= M/2 of one WMMA)
#define TROWS 13            // staged source rows: h-2 .. h+10
#define WSH   368           // padded ws plane height (46*8)
#define WSW   768           // padded ws plane width  (6*128)
#define WSP   ((size_t)WSH * WSW)

#define R_EARTH 6371000.0f
#define RINV    (1.0f / R_EARTH)
#define RINV2   (RINV * RINV)
#define R_GAS   287.0f
#define C_P     1004.0f
#define NU_     200000.0f

// ---------------- CDNA5 async global->LDS copy (probe-confirmed) -----------
#if defined(__has_builtin)
#  if __has_builtin(__builtin_amdgcn_global_load_async_to_lds_b32)
#    define HAVE_ASYNC32 1
#  else
#    define HAVE_ASYNC32 0
#  endif
#  if __has_builtin(__builtin_amdgcn_global_load_async_to_lds_b64)
#    define HAVE_ASYNC64 1
#  else
#    define HAVE_ASYNC64 0
#  endif
#else
#  define HAVE_ASYNC32 0
#  define HAVE_ASYNC64 0
#endif

typedef __attribute__((address_space(1))) int   g_i32;
typedef __attribute__((address_space(3))) int   l_i32;
typedef __attribute__((address_space(1))) int2v g_i64;
typedef __attribute__((address_space(3))) int2v l_i64;

__device__ __forceinline__ void cp_b32(float* l, const float* g) {
#if HAVE_ASYNC32
  __builtin_amdgcn_global_load_async_to_lds_b32((g_i32*)(g), (l_i32*)(l), 0, 0);
#else
  *l = *g;
#endif
}
__device__ __forceinline__ void cp_b64(float* l, const float* g) {
#if HAVE_ASYNC64
  __builtin_amdgcn_global_load_async_to_lds_b64((g_i64*)(g), (l_i64*)(l), 0, 0);
#else
  l[0] = g[0]; l[1] = g[1];
#endif
}
__device__ __forceinline__ void async_wait_all() {
#if HAVE_ASYNC32 || HAVE_ASYNC64
  asm volatile("s_wait_asynccnt 0x0" ::: "memory");
#endif
}

__device__ __forceinline__ int clampi(int x, int lo, int hi) {
  return x < lo ? lo : (x > hi ? hi : x);
}

// ---------------------------------------------------------------------------
// A fragment (16x32 bf16, one K=30 chunk). ISA layout: lanes 0-15: V0-3 ->
// K 0..7, V4-7 -> K 16..23; lanes 16-31: K 8..15 / 24..31.
// Row m = 2*hr + nc carries psiS[nc, hb+hr, dl, dw]*quad for taps whose
// source row sr = rowBase + K/5 has dl = sr - hr in [0,4]; else 0.
// Built ONCE per kernel: all masking happens here, not in the hot loop.
// ---------------------------------------------------------------------------
__device__ __forceinline__ bf16x16 build_A(const float* __restrict__ psi,
                                           const float* __restrict__ quad,
                                           int hb, int rowBase) {
  const int lane = threadIdx.x & 31;
  const int M    = lane & 15;
  const int hr   = M >> 1;
  const int nc   = M & 1;
  const int hi   = (lane >= 16) ? 1 : 0;
  bf16x16 a;
#pragma unroll
  for (int e = 0; e < 16; ++e) {
    const int v  = e >> 1, p = e & 1;
    const int Kk = (v < 4) ? ((hi ? 8 : 0) + 2 * v + p)
                           : ((hi ? 24 : 16) + 2 * (v - 4) + p);
    const int lr = Kk / 5, dw = Kk % 5;
    const int sr = rowBase + lr;
    const int dl = sr - hr;
    const int ho = hb + hr;
    float val = 0.0f;
    if (Kk < 30 && dl >= 0 && dl <= 4 && ho < HGT) {
      float ps = 0.0f;
#pragma unroll
      for (int kk = 0; kk < KKN; ++kk)    // psiS: pre-sum over kk
        ps += psi[(((kk * 2 + nc) * HGT + ho) * 5 + dl) * 5 + dw];
      val = ps * quad[clampi(hb - 2 + sr, 0, HGT - 1)];
    }
    a[e] = (__bf16)val;
  }
  return a;
}

// ---------------------------------------------------------------------------
// B fragment (32x16 bf16): lanes 0-15 K=0..7,16..23, lanes 16-31 K=8..15,24..31, N=lane%16.
// Unconditional fp32 LDS reads (always 4B aligned), bf16 convert at pack.
// `stride` = 1 (scalar tile) or 2 (interleaved uv tile), constant per call.
// ---------------------------------------------------------------------------
__device__ __forceinline__ bf16x16 build_B(const float* __restrict__ tp,
                                           int stride, int rowBase, int col) {
  const int kofs = ((threadIdx.x & 31) >= 16) ? 16 : 0;
  bf16x16 b;
#pragma unroll
  for (int e = 0; e < 16; ++e) {
    const int Kk = (e < 8) ? ((kofs >> 1) + e) : (16 + (kofs >> 1) + (e - 8));
    const int lr = Kk / 5, dw = Kk % 5;
    b[e] = (__bf16)tp[((rowBase + lr) * TCOLS + col + dw) * stride];
  }
  return b;
}

__device__ __forceinline__ floatx8 wmma0(bf16x16 a, bf16x16 b) {
  floatx8 c = {};
  return __builtin_amdgcn_wmma_f32_16x16x32_bf16(false, a, false, b,
                                                 (short)0, c, false, false);
}
__device__ __forceinline__ floatx8 wmmac(bf16x16 a, bf16x16 b, floatx8 c) {
  return __builtin_amdgcn_wmma_f32_16x16x32_bf16(false, a, false, b,
                                                 (short)0, c, false, false);
}

// ===========================================================================
// Stage 1: per level k, stencil planes of {u,v,T,q,ke} -> ws plane groups
// [0]=SG0(u) [1]=SG1(u) [2]=SG1(v) [3]=SG1(T) [4]=SG1(q) [5]=SG1(ke),
// each group KLEV padded 368x768 planes. Stores are fully unguarded.
// ===========================================================================
__global__ __launch_bounds__(256) void pe_stage1_gather_wmma(
    const float* __restrict__ uv, const float* __restrict__ T,
    const float* __restrict__ q, const float* __restrict__ psi,
    const float* __restrict__ quad, float* __restrict__ ws) {
  const int hb = blockIdx.y * HROWS;
  const int w0 = blockIdx.x * WTILE;

  __shared__ float uvt[TROWS * TCOLS * 2];   // interleaved u,v  13.7 KB
  __shared__ float Tt [TROWS * TCOLS];       // 6.9 KB
  __shared__ float qt [TROWS * TCOLS];       // 6.9 KB
  __shared__ float ket[TROWS * TCOLS];       // 6.9 KB

  __builtin_prefetch(&psi[hb * 25], 0, 0);   // global_prefetch_b8

  const bf16x16 A1 = build_A(psi, quad, hb, 0);   // src rows 0..5
  const bf16x16 A2 = build_A(psi, quad, hb, 6);   // src rows 6..11

  const int wv     = threadIdx.x >> 5;
  const int lane   = threadIdx.x & 31;
  const int col    = wv * 16 + (lane & 15);       // local column 0..127
  const int w      = w0 + col;
  const int hrBase = (lane >= 16) ? 4 : 0;

  for (int k = 0; k < KLEV; ++k) {
    __syncthreads();                         // tiles free from previous level
    // ---- async-stage level k (lat clipped, lon wrapped) ----
    for (int idx = threadIdx.x; idx < TROWS * TCOLS; idx += 256) {
      const int r   = idx / TCOLS;
      const int c   = idx % TCOLS;
      const int row = clampi(hb - 2 + r, 0, HGT - 1);
      const int cg  = (w0 - 2 + c + WID) % WID;
      const size_t base = ((size_t)(k * HGT + row)) * WID + cg;
      cp_b64(&uvt[idx * 2], &uv[base * 2]);
      cp_b32(&Tt[idx], &T[base]);
      cp_b32(&qt[idx], &q[base]);
    }
    async_wait_all();
    // ke derived in-LDS from entries this thread itself staged
    for (int idx = threadIdx.x; idx < TROWS * TCOLS; idx += 256) {
      const float uu = uvt[idx * 2], vv = uvt[idx * 2 + 1];
      ket[idx] = 0.5f * (uu * uu + vv * vv);
    }
    __syncthreads();

    for (int f = 0; f < 5; ++f) {            // 0:u 1:v 2:T 3:q 4:ke
      const float* tp = (f == 0) ? uvt : (f == 1) ? uvt + 1
                      : (f == 2) ? Tt : (f == 3) ? qt : ket;
      const int st = (f < 2) ? 2 : 1;
      floatx8 d = wmma0(A1, build_B(tp, st, 0, col));
      d = wmmac(A2, build_B(tp, st, 6, col), d);

      const int g1 = (f == 0) ? 1 : (f + 1);             // nc=1 plane group
      const size_t ob = (size_t)(g1 * KLEV + k) * WSP
                      + (size_t)(hb + hrBase) * WSW + w;
#pragma unroll
      for (int j = 0; j < 4; ++j) ws[ob + (size_t)j * WSW] = d[2 * j + 1];
      if (f == 0) {                                      // SG0(u) as well
        const size_t ob0 = (size_t)(0 * KLEV + k) * WSP
                         + (size_t)(hb + hrBase) * WSW + w;
#pragma unroll
        for (int j = 0; j < 4; ++j) ws[ob0 + (size_t)j * WSW] = d[2 * j];
      }
    }
  }
}

// ===========================================================================
// Stage 2: lap(s) = SG1(SG1(s))/R^2 for s in {T,q,u,v}, fused with all
// pointwise vertical physics. Each lane owns 4 (latitude,column) positions;
// omega / pressure-gradient prefix accumulators live in registers across k.
// ===========================================================================
__global__ __launch_bounds__(256) void pe_stage2_lap_assemble(
    const float* __restrict__ uv, const float* __restrict__ T,
    const float* __restrict__ psi, const float* __restrict__ quad,
    const float* __restrict__ f_cor, const float* __restrict__ p_levels,
    const float* __restrict__ delta_p, const float* __restrict__ ws,
    float* __restrict__ out) {
  const int hb = blockIdx.y * HROWS;
  const int w0 = blockIdx.x * WTILE;
  const size_t HW    = (size_t)HGT * WID;
  const size_t OFF_T = 2 * (size_t)KLEV * HW;
  const size_t OFF_Q = OFF_T + (size_t)KLEV * HW;

  __shared__ float gt[4 * TROWS * TCOLS];    // sg-planes {T,q,u,v}, 27.5 KB

  const bf16x16 A1 = build_A(psi, quad, hb, 0);
  const bf16x16 A2 = build_A(psi, quad, hb, 6);

  const int wv     = threadIdx.x >> 5;
  const int lane   = threadIdx.x & 31;
  const int col    = wv * 16 + (lane & 15);
  const int w      = w0 + col;
  const int hrBase = (lane >= 16) ? 4 : 0;

  float cumPG[4] = {0, 0, 0, 0};   // index-constant via unrolled j loops
  float cum[4]   = {0, 0, 0, 0};
  float prv[4]   = {0, 0, 0, 0};

  for (int k = 0; k < KLEV; ++k) {
    __syncthreads();
    // ---- async-stage the four SG1 planes for level k (raw, unscaled) ----
    for (int idx = threadIdx.x; idx < 4 * TROWS * TCOLS; idx += 256) {
      const int c   = idx % TCOLS;
      const int t   = idx / TCOLS;
      const int r   = t % TROWS;
      const int f2  = t / TROWS;                        // 0:T 1:q 2:u 3:v
      const int g   = (f2 == 0) ? 3 : (f2 == 1) ? 4 : (f2 == 2) ? 1 : 2;
      const int row = clampi(hb - 2 + r, 0, HGT - 1);
      const int cg  = (w0 - 2 + c + WID) % WID;
      cp_b32(&gt[idx], &ws[(size_t)(g * KLEV + k) * WSP
                           + (size_t)row * WSW + cg]);
    }
    async_wait_all();
    __syncthreads();

    // 4 chained-WMMA Laplacians for level k (uniform control flow)
    const float* tp0 = &gt[0 * TROWS * TCOLS];
    const float* tp1 = &gt[1 * TROWS * TCOLS];
    const float* tp2 = &gt[2 * TROWS * TCOLS];
    const float* tp3 = &gt[3 * TROWS * TCOLS];
    floatx8 dT_ = wmmac(A2, build_B(tp0, 1, 6, col),
                        wmma0(A1, build_B(tp0, 1, 0, col)));
    floatx8 dq_ = wmmac(A2, build_B(tp1, 1, 6, col),
                        wmma0(A1, build_B(tp1, 1, 0, col)));
    floatx8 du_ = wmmac(A2, build_B(tp2, 1, 6, col),
                        wmma0(A1, build_B(tp2, 1, 0, col)));
    floatx8 dv_ = wmmac(A2, build_B(tp3, 1, 6, col),
                        wmma0(A1, build_B(tp3, 1, 0, col)));

    // uniform per-level scalars (s_load'ed, L2-resident)
    const float pk   = p_levels[k];
    const float pkp  = p_levels[k < KLEV - 1 ? k + 1 : k];
    const float pkm  = p_levels[k > 0 ? k - 1 : k];
    const float dpk  = delta_p[k];
    const float lprk = (k < KLEV - 1) ? R_GAS * __logf(pk / pkp) : 0.0f;

#pragma unroll
    for (int j = 0; j < 4; ++j) {
      const int ho = hb + hrBase + j;
      if (ho < HGT && w < WID) {
        const size_t o   = (size_t)ho * WSW + w;        // padded ws row
        const size_t pos = (size_t)(k * HGT + ho) * WID + w;
        const float u  = uv[2 * pos], v = uv[2 * pos + 1];
        const float Tk = T[pos];
        const float sgu0  = ws[(0 * KLEV + k) * WSP + o];
        const float sgv1  = ws[(2 * KLEV + k) * WSP + o];
        const float sgT1  = ws[(3 * KLEV + k) * WSP + o];
        const float sgq1  = ws[(4 * KLEV + k) * WSP + o];
        const float sgke1 = ws[(5 * KLEV + k) * WSP + o];
        const float lapT = dT_[2 * j + 1] * RINV2;
        const float lapq = dq_[2 * j + 1] * RINV2;
        const float lapu = du_[2 * j + 1] * RINV2;
        const float lapv = dv_[2 * j + 1] * RINV2;

        const float coef = sgu0 * RINV - f_cor[ho];
        const float yv   = -RINV * (cumPG[j] + sgke1);
        cumPG[j] += lprk * sgT1;                        // prefix for k'>k
        cum[j]   += (sgv1 * RINV) * dpk;                // omega cumsum
        const float omega = 0.5f * (prv[j] + cum[j]);
        prv[j] = cum[j];

        float dTdp;
        if      (k == 0)        dTdp = (T[pos + HW] - Tk) / (pkp - pk);
        else if (k == KLEV - 1) dTdp = (Tk - T[pos - HW]) / (pk - pkm);
        else                    dTdp = (T[pos + HW] - T[pos - HW]) / (pkp - pkm);

        const float adv = -(v * RINV);
        out[2 * pos]     = -coef * v + NU_ * lapu;
        out[2 * pos + 1] =  coef * u + yv + NU_ * lapv;
        out[OFF_T + pos] = adv * sgT1
                         + omega * (R_GAS / C_P * Tk / pk - dTdp)
                         + NU_ * lapT;
        out[OFF_Q + pos] = adv * sgq1 + NU_ * lapq;
      }
    }
  }
}

// ===========================================================================
extern "C" void kernel_launch(void* const* d_in, const int* in_sizes, int n_in,
                              void* d_out, int out_size, void* d_ws, size_t ws_size,
                              hipStream_t stream) {
  (void)in_sizes; (void)n_in; (void)out_size; (void)ws_size;
  const float* uv       = (const float*)d_in[0];
  const float* T        = (const float*)d_in[1];
  const float* q        = (const float*)d_in[2];
  const float* psi      = (const float*)d_in[3];
  const float* quad     = (const float*)d_in[4];
  const float* f_cor    = (const float*)d_in[5];
  const float* p_levels = (const float*)d_in[6];
  const float* delta_p  = (const float*)d_in[7];
  float* ws  = (float*)d_ws;   // 48 padded fp32 planes (368x768) ~54 MB
  float* out = (float*)d_out;

  dim3 grid((WID + WTILE - 1) / WTILE, (HGT + HROWS - 1) / HROWS);  // 6 x 46
  pe_stage1_gather_wmma<<<grid, 256, 0, stream>>>(uv, T, q, psi, quad, ws);
  pe_stage2_lap_assemble<<<grid, 256, 0, stream>>>(uv, T, psi, quad, f_cor,
                                                   p_levels, delta_p, ws, out);
}
